// LocalFeatureAggregation_19602230739187
// MI455X (gfx1250) — matrix operand order repaired and checked by reading, not compiled, not run on hardware
//
#include <hip/hip_runtime.h>
#include <hip/hip_bf16.h>

typedef __attribute__((ext_vector_type(16))) _Float16 v16h;
typedef __attribute__((ext_vector_type(8)))  float    v8f;
typedef __attribute__((ext_vector_type(4)))  float    v4f;

#define B_      2
#define N_      65536
#define KNB_    16
#define CR_     10
#define CIN_    64
#define CNB_    64
#define CCAT_   128
#define COUT_   128
#define BN_EPS  1e-5f
#define NEG_SLOPE 0.2f

#define THREADS 256
#define PPB     128      /* points per block; 131072/128 = 1024 blocks */

__device__ __forceinline__ v8f wmma16(v16h a, v16h b, v8f c) {
    // v_wmma_f32_16x16x32_f16: (neg_a, A, neg_b, B, c_mod, C, reuse_a, reuse_b)
    return __builtin_amdgcn_wmma_f32_16x16x32_f16(false, a, false, b, (short)0, c, false, false);
}

// A-matrix (16xK) tile built from f32 LDS with cvt->f16.
// Layout per ISA 7.12.2: lane L covers M = L&15; VGPR v holds K pair
//   k = (v&4 ? 16:0) + (v&3)*2 + (L>=16 ? 8:0)
__device__ __forceinline__ v16h load_AtileF32(const float* S, int ld, int kbase, int lane) {
    v16h a;
    const int m  = lane & 15;
    const int hk = (lane >= 16) ? 8 : 0;
    const float* row = S + m * ld;
#pragma unroll
    for (int v = 0; v < 8; ++v) {
        int k = kbase + ((v & 4) ? 16 : 0) + (v & 3) * 2 + hk;
        a[2 * v]     = (_Float16)row[k];
        a[2 * v + 1] = (_Float16)row[k + 1];
    }
    return a;
}

// B-matrix (32x16, K x N) tile from a row-major f32 weight:
// VGPR v holds K pair k = (v&4 ? 16:0) + (v&3)*2 + (L>=16 ? 8:0), matching the A tile.
__device__ __forceinline__ v16h load_Btile(const float* W, int ld, int kbase, int nbase, int lane) {
    v16h b;
    const int n  = nbase + (lane & 15);
    const int kh = (lane >= 16) ? 8 : 0;
#pragma unroll
    for (int v = 0; v < 8; ++v) {
        int k0 = kbase + ((v & 4) ? 16 : 0) + (v & 3) * 2 + kh;
        b[2 * v]     = (_Float16)W[(long)k0 * ld + n];
        b[2 * v + 1] = (_Float16)W[(long)(k0 + 1) * ld + n];
    }
    return b;
}

// CDNA5 async copy: global -> LDS, 16B per active lane, tracked by ASYNCcnt.
__device__ __forceinline__ void async_b128(const void* lds, const void* g) {
    unsigned off = (unsigned)(unsigned long long)lds;   // low 32 bits = LDS offset
    asm volatile("global_load_async_to_lds_b128 %0, %1, off"
                 :: "v"(off), "v"(g) : "memory");
}
__device__ __forceinline__ void async_wait0() {
    asm volatile("s_wait_asynccnt 0x0" ::: "memory");
}

__global__ __launch_bounds__(THREADS)
void lfa_kernel(const float* __restrict__ feature,
                const float* __restrict__ raw,        // [B,N,K,Cr]
                const int*   __restrict__ nidx,       // [B,N,K]
                const float* __restrict__ W_nb,
                const float* __restrict__ b_nb, const float* __restrict__ g_nb,
                const float* __restrict__ be_nb, const float* __restrict__ m_nb,
                const float* __restrict__ v_nb,
                const float* __restrict__ W_attn,
                const float* __restrict__ W_out,
                const float* __restrict__ b_out, const float* __restrict__ g_out,
                const float* __restrict__ be_out, const float* __restrict__ m_out,
                const float* __restrict__ v_out,
                const float* __restrict__ W_sc,
                const float* __restrict__ b_sc, const float* __restrict__ g_sc,
                const float* __restrict__ be_sc, const float* __restrict__ m_sc,
                const float* __restrict__ v_sc,
                float* __restrict__ out)
{
    __shared__ __align__(16) float feaS[2][KNB_][CCAT_];   // double-buffered fea (f32)
    __shared__ __align__(16) float pooledS[2][16][CCAT_];  // per-batch pooled vectors
    __shared__ __align__(16) float featS[2][16][CIN_];     // per-batch own features
    __shared__ __align__(16) float rawB[2][KNB_ * CR_];
    __shared__ __align__(16) int   idxB[2][KNB_];
    __shared__ __align__(16) float wnbS[CR_ * CNB_];
    __shared__ __align__(16) float nbScale[CNB_], nbShift[CNB_];

    const int tid  = threadIdx.x;
    const int lane = tid & 31;
    const int wave = tid >> 5;
    const int ntile = wave;
    const int base = blockIdx.x * PPB;

    // ---- one-time block setup ------------------------------------------
    for (int i = tid; i < CR_ * CNB_; i += THREADS) wnbS[i] = W_nb[i];
    if (tid < CNB_) {
        float sc = g_nb[tid] * rsqrtf(v_nb[tid] + BN_EPS);
        nbScale[tid] = sc;
        nbShift[tid] = be_nb[tid] + (b_nb[tid] - m_nb[tid]) * sc;
    }

    v16h Wa[4], Wo[4], Ws[2];
#pragma unroll
    for (int kc = 0; kc < 4; ++kc) Wa[kc] = load_Btile(W_attn, CCAT_, kc * 32, ntile * 16, lane);
#pragma unroll
    for (int kc = 0; kc < 4; ++kc) Wo[kc] = load_Btile(W_out, COUT_, kc * 32, ntile * 16, lane);
#pragma unroll
    for (int kc = 0; kc < 2; ++kc) Ws[kc] = load_Btile(W_sc, COUT_, kc * 32, ntile * 16, lane);

    const int ccol = ntile * 16 + (lane & 15);
    const float scO = g_out[ccol] * rsqrtf(v_out[ccol] + BN_EPS);
    const float shO = be_out[ccol] + (b_out[ccol] - m_out[ccol]) * scO;
    const float scS = g_sc[ccol] * rsqrtf(v_sc[ccol] + BN_EPS);
    const float shS = be_sc[ccol] + (b_sc[ccol] - m_sc[ccol]) * scS;

    // ---- pipeline stages (q = local point index 0..PPB-1) ---------------
    auto stage_raw = [&](int q) {
        if (tid < (KNB_ * CR_) / 4)
            async_b128(&rawB[q & 1][tid * 4],
                       raw + (long)(base + q) * (KNB_ * CR_) + tid * 4);
    };
    auto stage_idx = [&](int q) {
        if (tid < KNB_ / 4)
            async_b128(&idxB[q & 1][tid * 4],
                       nidx + (long)(base + q) * KNB_ + tid * 4);
    };
    auto stage_feat = [&](int q) {
        if (tid < CIN_ / 4)
            async_b128(&featS[(q >> 4) & 1][q & 15][tid * 4],
                       feature + (long)(base + q) * CIN_ + tid * 4);
    };
    auto gather = [&](int q) {   // 256 lanes: one b128 per (neighbor k, 4 channels)
        int k  = tid >> 4;
        int c4 = (tid & 15) * 4;
        int bidx = (base + q) / N_;
        int gi = idxB[q & 1][k];
        async_b128(&feaS[q & 1][k][c4],
                   feature + ((long)bidx * N_ + gi) * CIN_ + c4);
    };
    auto nbmlp = [&](int q) {    // Linear(10->64)+BN+lrelu, 1 neighbor x 4 ch / thread
        int k  = tid >> 4;
        int c4 = (tid & 15) * 4;
        const float* rrow = &rawB[q & 1][k * CR_];
        v4f acc = {};
#pragma unroll
        for (int j = 0; j < CR_; ++j) {
            v4f w = *(const v4f*)&wnbS[j * CNB_ + c4];
            float rj = rrow[j];
            acc[0] += rj * w[0]; acc[1] += rj * w[1];
            acc[2] += rj * w[2]; acc[3] += rj * w[3];
        }
        v4f sc = *(const v4f*)&nbScale[c4];
        v4f sh = *(const v4f*)&nbShift[c4];
#pragma unroll
        for (int t = 0; t < 4; ++t) {
            float vv = acc[t] * sc[t] + sh[t];
            acc[t] = (vv >= 0.f) ? vv : NEG_SLOPE * vv;
        }
        *(v4f*)&feaS[q & 1][k][CNB_ + c4] = acc;
    };
    auto attn_pool = [&](int q) {
        const float* F = &feaS[q & 1][0][0];
        v8f acc = {};
#pragma unroll
        for (int kc = 0; kc < 4; ++kc)
            acc = wmma16(load_AtileF32(F, CCAT_, kc * 32, lane), Wa[kc], acc);
        // softmax over 16 neighbors (M axis): 8 regs + lane^16 half
        float mx = acc[0];
#pragma unroll
        for (int r = 1; r < 8; ++r) mx = fmaxf(mx, acc[r]);
        mx = fmaxf(mx, __shfl_xor(mx, 16));
        float ex[8], ssum = 0.f;
#pragma unroll
        for (int r = 0; r < 8; ++r) { ex[r] = __expf(acc[r] - mx); ssum += ex[r]; }
        ssum += __shfl_xor(ssum, 16);
        const float inv = 1.0f / ssum;
        const int mh = (lane >= 16) ? 8 : 0;
        float part = 0.f;
#pragma unroll
        for (int r = 0; r < 8; ++r)
            part += ex[r] * inv * F[(r + mh) * CCAT_ + ccol];
        part += __shfl_xor(part, 16);
        if (lane < 16) pooledS[(q >> 4) & 1][q & 15][ccol] = part;
    };
    auto epilogue = [&](int qb) {  // batch of 16 points: out + sc GEMMs, fused BN+lrelu
        const int buf = qb & 1;
        v8f accO = {}, accS = {};
#pragma unroll
        for (int kc = 0; kc < 4; ++kc)
            accO = wmma16(load_AtileF32(&pooledS[buf][0][0], CCAT_, kc * 32, lane), Wo[kc], accO);
#pragma unroll
        for (int kc = 0; kc < 2; ++kc)
            accS = wmma16(load_AtileF32(&featS[buf][0][0], CIN_, kc * 32, lane), Ws[kc], accS);
        const int mh = (lane >= 16) ? 8 : 0;
        const int pbase = base + qb * 16;
#pragma unroll
        for (int r = 0; r < 8; ++r) {
            float v = accO[r] * scO + shO + accS[r] * scS + shS;
            v = (v >= 0.f) ? v : NEG_SLOPE * v;
            out[(long)(pbase + r + mh) * COUT_ + ccol] = v;
        }
    };

    // ---- prologue: fill pipeline (depth 2) ------------------------------
    stage_idx(0); stage_raw(0);
    async_wait0();
    __syncthreads();
    gather(0); stage_feat(0); stage_idx(1); stage_raw(1);
    nbmlp(0);
    async_wait0();
    __syncthreads();

    // ---- steady state: 1 barrier per point ------------------------------
    for (int s = 0; s < PPB; ++s) {
        if (s + 2 < PPB) { stage_idx(s + 2); stage_raw(s + 2); }
        if (s + 1 < PPB) { gather(s + 1); stage_feat(s + 1); nbmlp(s + 1); }
        attn_pool(s);
        async_wait0();
        __syncthreads();
        if ((s & 15) == 15) epilogue(s >> 4);
    }
}

extern "C" void kernel_launch(void* const* d_in, const int* in_sizes, int n_in,
                              void* d_out, int out_size, void* d_ws, size_t ws_size,
                              hipStream_t stream) {
    const float* feature = (const float*)d_in[1];
    const float* raw     = (const float*)d_in[2];
    const int*   nidx    = (const int*)  d_in[3];
    const float* W_nb    = (const float*)d_in[4];
    const float* b_nb    = (const float*)d_in[5];
    const float* g_nb    = (const float*)d_in[6];
    const float* be_nb   = (const float*)d_in[7];
    const float* m_nb    = (const float*)d_in[8];
    const float* v_nb    = (const float*)d_in[9];
    const float* W_attn  = (const float*)d_in[10];
    const float* W_out   = (const float*)d_in[11];
    const float* b_out   = (const float*)d_in[12];
    const float* g_out   = (const float*)d_in[13];
    const float* be_out  = (const float*)d_in[14];
    const float* m_out   = (const float*)d_in[15];
    const float* v_out   = (const float*)d_in[16];
    const float* W_sc    = (const float*)d_in[17];
    const float* b_sc    = (const float*)d_in[18];
    const float* g_sc    = (const float*)d_in[19];
    const float* be_sc   = (const float*)d_in[20];
    const float* m_sc    = (const float*)d_in[21];
    const float* v_sc    = (const float*)d_in[22];
    float* out = (float*)d_out;

    dim3 grid((B_ * N_) / PPB), block(THREADS);
    hipLaunchKernelGGL(lfa_kernel, grid, block, 0, stream,
                       feature, raw, nidx,
                       W_nb, b_nb, g_nb, be_nb, m_nb, v_nb,
                       W_attn,
                       W_out, b_out, g_out, be_out, m_out, v_out,
                       W_sc, b_sc, g_sc, be_sc, m_sc, v_sc,
                       out);
}
